// ElectronicSchnet_25177098289470
// MI455X (gfx1250) — matrix operand order repaired and checked by reading, not compiled, not run on hardware
//
#include <hip/hip_runtime.h>

// ---------------------------------------------------------------------------
// ElectronicSchnet on MI455X (gfx1250, wave32, WMMA).
//
//  * bf16 everywhere through v_wmma_f32_16x16x32_bf16 (f32 accum)
//  * weights pre-swizzled in global memory into WMMA B-fragment order so
//    every B fragment is 2x ds_load_b128
//  * A fragments loaded as 2x ds_load_b128 (contiguous 16B runs per lane)
//  * bulk LDS staging via CDNA5 async-to-LDS (global_load_async_to_lds_b128,
//    ASYNCcnt), overlapped with f32->bf16 VGPR-path conversion staging
//  * bias + shifted-softplus fused between layers via LDS restage
//  * pair epilogue (x h[j], diag mask, segment-sum) fused on WMMA accums
// ---------------------------------------------------------------------------

typedef __attribute__((ext_vector_type(16))) __bf16 v16bf;
typedef __attribute__((ext_vector_type(8)))  __bf16 bf16x8;
typedef __attribute__((ext_vector_type(8)))  float  v8f;

#define LOG2F_ 0.69314718055994530942f

#define NB    1024
#define NE    32
#define NN    4
#define BAS   32
#define KER   64
#define EMB   128
#define MW    45         // -> pad 48 (N), 64 (K)
#define MG    91         // -> pad 96

__device__ __forceinline__ float ssp(float x) {
  return fmaxf(x, 0.f) + __logf(1.f + __expf(-fabsf(x))) - LOG2F_;
}

__device__ __forceinline__ v8f wmma_bf16(v16bf a, v16bf b, v8f c) {
  return __builtin_amdgcn_wmma_f32_16x16x32_bf16(false, a, false, b,
                                                 (short)0, c, false, false);
}

__device__ __forceinline__ v16bf cat8(bf16x8 lo, bf16x8 hi) {
  return __builtin_shufflevector(lo, hi, 0, 1, 2, 3, 4, 5, 6, 7,
                                         8, 9, 10, 11, 12, 13, 14, 15);
}

// ---- CDNA5 async copy: global -> LDS, 16B per lane, tracked by ASYNCcnt ----
// Generic LDS pointer low 32 bits are the LDS byte address (ISA 10.2 aperture
// rules).
__device__ __forceinline__ void async_cp16(void* lds, const void* g) {
  unsigned l = (unsigned)(unsigned long long)lds;
  asm volatile("global_load_async_to_lds_b128 %0, %1, off"
               :: "v"(l), "v"(g) : "memory");
}
__device__ __forceinline__ void wait_async0() {
  asm volatile("s_wait_asynccnt 0x0" ::: "memory");
}

// A fragment (16x32 bf16 tile, row-major LDS). Per ISA 7.12.2 the per-lane
// data is two contiguous 8-element runs: k in [half*8,+8) and [16+half*8,+8).
__device__ __forceinline__ v16bf load_A_frag(const __bf16* lds, int row0, int k0,
                                             int stride) {
  int lane = threadIdx.x & 31;
  int m = lane & 15, half = lane >> 4;
  const __bf16* p = lds + (row0 + m) * stride + k0 + half * 8;
  bf16x8 lo = *(const bf16x8*)(p);
  bf16x8 hi = *(const bf16x8*)(p + 16);
  return cat8(lo, hi);
}

// B fragment from pre-swizzled weights: layout [tile][lane][16] contiguous.
__device__ __forceinline__ v16bf load_B_frag_swz(const __bf16* lds, int tile) {
  int lane = threadIdx.x & 31;
  const __bf16* p = lds + ((tile << 5) + lane) * 16;
  bf16x8 lo = *(const bf16x8*)(p);
  bf16x8 hi = *(const bf16x8*)(p + 8);
  return cat8(lo, hi);
}

// 8 f32 (two float4 loads) -> bf16x8 store (one b128 store)
__device__ __forceinline__ void cvt8(__bf16* dst, const float* __restrict__ src) {
  float4 a = ((const float4*)src)[0];
  float4 b = ((const float4*)src)[1];
  bf16x8 o;
  o[0] = (__bf16)a.x; o[1] = (__bf16)a.y; o[2] = (__bf16)a.z; o[3] = (__bf16)a.w;
  o[4] = (__bf16)b.x; o[5] = (__bf16)b.y; o[6] = (__bf16)b.z; o[7] = (__bf16)b.w;
  *(bf16x8*)dst = o;
}

// ---------------------------------------------------------------------------
// Prep kernels
// ---------------------------------------------------------------------------
__global__ void k_cvt8(const float* __restrict__ src, __bf16* __restrict__ dst,
                       size_t n8) {
  size_t i = (size_t)blockIdx.x * blockDim.x + threadIdx.x;
  if (i < n8) cvt8(dst + i * 8, src + i * 8);
}

// convert + zero-pad + swizzle into B-fragment order. Launch <<<dK, dN>>> so
// index decomposition is shifts/masks only (no idiv VALU expansion).
// dst[(((ks*nnt + nt)*32 + lane)*16) + t], lane = (kg<<4)|(n&15), kg = (k>>3)&1, t = (k&7)|(((k&31)>>4)<<3)
__global__ void k_swz_w(const float* __restrict__ src, __bf16* __restrict__ dst,
                        int sR, int sC) {
  int k = blockIdx.x, n = threadIdx.x, dN = blockDim.x;
  float v = (k < sR && n < sC) ? src[k * sC + n] : 0.f;
  int nnt = dN >> 4;
  int ks = k >> 5, kk = k & 31, kg = (kk >> 3) & 1, t = (kk & 7) | ((kk >> 4) << 3);
  int nt = n >> 4, lane = (kg << 4) | (n & 15);
  dst[((((ks * nnt + nt) << 5) + lane) << 4) + t] = (__bf16)v;
}

__global__ void k_pad_f32(const float* __restrict__ src, float* __restrict__ dst,
                          int sN, int dN) {
  int i = blockIdx.x * blockDim.x + threadIdx.x;
  if (i >= dN) return;
  dst[i] = (i < sN) ? src[i] : 0.f;
}

__global__ void k_xinit(const float4* __restrict__ X4, float4* __restrict__ x4) {
  size_t idx = (size_t)blockIdx.x * blockDim.x + threadIdx.x;
  if (idx < (size_t)NB * NE * EMB / 4) x4[idx] = X4[idx & (NE * EMB / 4 - 1)];
}

// ---------------------------------------------------------------------------
// h = x @ hW : [B*32,128] x [128,64] -> [B*32,64]. 8 waves x 16 rows / block.
// ---------------------------------------------------------------------------
__global__ __launch_bounds__(256) void k_h(const float* __restrict__ x,
                                           const __bf16* __restrict__ hWswz,
                                           float* __restrict__ h) {
  __shared__ __align__(16) __bf16 sW[128 * 64];      // 16 tiles, swizzled
  __shared__ __align__(16) __bf16 sA[8][16 * 128];
  int tid = threadIdx.x, lane = tid & 31, wave = tid >> 5;

  // async: weights straight to LDS; overlap with x f32->bf16 conversion
  for (int t = tid; t < 1024; t += 256)
    async_cp16((char*)sW + t * 16, (const char*)hWswz + t * 16);
  int rowB = blockIdx.x * 128;
  const float* xs = x + (size_t)rowB * EMB;
  for (int c = tid * 8; c < 128 * 128; c += 256 * 8)
    cvt8(&sA[0][0] + c, xs + c);
  wait_async0();
  __syncthreads();

  v8f acc[4] = {};
#pragma unroll
  for (int ks = 0; ks < 4; ++ks) {
    v16bf a = load_A_frag(sA[wave], 0, ks * 32, 128);
#pragma unroll
    for (int nt = 0; nt < 4; ++nt)
      acc[nt] = wmma_bf16(a, load_B_frag_swz(sW, ks * 4 + nt), acc[nt]);
  }
  int nb = lane & 15, mh8 = (lane >> 4) << 3;
  float* hp = h + (size_t)(rowB + wave * 16) * KER;
#pragma unroll
  for (int nt = 0; nt < 4; ++nt)
#pragma unroll
    for (int r = 0; r < 8; ++r)
      hp[(mh8 + r) * KER + nt * 16 + nb] = acc[nt][r];
}

// ---------------------------------------------------------------------------
// Pair messages. Block = 256 thr = 8 waves = 4 (b,i) units x 2 j-halves.
// ---------------------------------------------------------------------------
__global__ __launch_bounds__(256) void k_pair(
    const __bf16* __restrict__ edges,  // [B,32,32,32] bf16
    const __bf16* __restrict__ w1p,    // [2][3 tiles]   swizzled (same, anti)
    const float* __restrict__ b1p,     // [2][48]
    const __bf16* __restrict__ w2p,    // [2][8 tiles]   swizzled
    const float* __restrict__ h,       // [B,32,64]
    float* __restrict__ z_same, float* __restrict__ z_anti) {
  __shared__ __align__(16) __bf16 sA[4][32 * 32];
  __shared__ __align__(16) __bf16 sW1[2][32 * 48];
  __shared__ __align__(16) __bf16 sW2[2][64 * 64];
  __shared__ __align__(16) float  sB1[2][48];
  __shared__ __align__(16) __bf16 sAct[8][16 * 64];

  int tid = threadIdx.x, lane = tid & 31, wave = tid >> 5;
  int unit = wave >> 1, jhalf = wave & 1;
  int bi0 = blockIdx.x * 4;

  // async staging: edges tile first (streamed, longest latency), then weights
  for (int t = tid; t < 512; t += 256)
    async_cp16((char*)sA + t * 16, (const char*)(edges + (size_t)bi0 * 1024) + t * 16);
  for (int t = tid; t < 384; t += 256)
    async_cp16((char*)sW1 + t * 16, (const char*)w1p + t * 16);
  for (int t = tid; t < 1024; t += 256)
    async_cp16((char*)sW2 + t * 16, (const char*)w2p + t * 16);
  if (tid < 24) async_cp16((char*)sB1 + tid * 16, (const char*)b1p + tid * 16);
  wait_async0();
  __syncthreads();

  int bi = bi0 + unit, b = bi >> 5, i = bi & 31;
  int sel = jhalf ^ (i >> 4);  // 0 = same-spin block, 1 = anti
  int nb = lane & 15, mh8 = (lane >> 4) << 3;

  // ---- layer 1 (K=32, N=48: 3 tiles) ----
  v16bf a1 = load_A_frag(sA[unit], jhalf * 16, 0, 32);
#pragma unroll
  for (int nt = 0; nt < 3; ++nt) {
    v8f z = {};
    v8f acc = wmma_bf16(a1, load_B_frag_swz(sW1[sel], nt), z);
#pragma unroll
    for (int r = 0; r < 8; ++r) {
      int n = nt * 16 + nb;
      sAct[wave][(mh8 + r) * 64 + n] = (__bf16)ssp(acc[r] + sB1[sel][n]);
    }
  }
  {  // zero K-pad cols 48..63: one b128 store per lane
    bf16x8 z = {};
    *(bf16x8*)&sAct[wave][(lane >> 1) * 64 + 48 + ((lane & 1) << 3)] = z;
  }
  // wave-private LDS region; DS ops are in-order per wave -> no barrier

  // ---- layer 2 (K=64 pad, N=64: 2x4 tiles) ----
  v8f acc2[4] = {};
#pragma unroll
  for (int ks = 0; ks < 2; ++ks) {
    v16bf a2 = load_A_frag(sAct[wave], 0, ks * 32, 64);
#pragma unroll
    for (int nt = 0; nt < 4; ++nt)
      acc2[nt] = wmma_bf16(a2, load_B_frag_swz(sW2[sel], ks * 4 + nt), acc2[nt]);
  }

  // ---- epilogue: * h[b,j,:], mask diagonal, reduce over j ----
  const float* hb = h + (size_t)b * NE * KER;
  float* zout = ((sel == 0) ? z_same : z_anti) + ((size_t)b * NE + i) * KER;
#pragma unroll
  for (int nt = 0; nt < 4; ++nt) {
    int n = nt * 16 + nb;
    float s = 0.f;
#pragma unroll
    for (int r = 0; r < 8; ++r) {
      int j = jhalf * 16 + mh8 + r;
      float v = acc2[nt][r] * hb[j * KER + n];
      s += (j == i) ? 0.f : v;
    }
    s += __shfl_xor(s, 16, 32);
    if (lane < 16) zout[n] = s;
  }
}

// ---------------------------------------------------------------------------
// Nuclear messages. Rows = B*32*4; 8 waves x 16 rows / block.
// ---------------------------------------------------------------------------
__global__ __launch_bounds__(256) void k_nuc(
    const __bf16* __restrict__ en,   // [B*32*4][32] bf16
    const __bf16* __restrict__ w1,   // swizzled [3 tiles]
    const float* __restrict__ b1,    // [48]
    const __bf16* __restrict__ w2,   // swizzled [8 tiles]
    const float* __restrict__ Y,     // [4*64] f32
    float* __restrict__ z_nuc) {
  __shared__ __align__(16) __bf16 sW1[32 * 48];
  __shared__ __align__(16) __bf16 sW2[64 * 64];
  __shared__ __align__(16) float  sB1[48];
  __shared__ __align__(16) float  sY[4 * 64];
  __shared__ __align__(16) __bf16 sA[8][16 * 32];
  __shared__ __align__(16) __bf16 sAct[8][16 * 64];

  int tid = threadIdx.x, lane = tid & 31, wave = tid >> 5;
  int row0 = blockIdx.x * 128;
  for (int t = tid; t < 512; t += 256)
    async_cp16((char*)sA + t * 16, (const char*)(en + (size_t)row0 * BAS) + t * 16);
  for (int t = tid; t < 192; t += 256)
    async_cp16((char*)sW1 + t * 16, (const char*)w1 + t * 16);
  for (int t = tid; t < 512; t += 256)
    async_cp16((char*)sW2 + t * 16, (const char*)w2 + t * 16);
  if (tid < 12) async_cp16((char*)sB1 + tid * 16, (const char*)b1 + tid * 16);
  if (tid < 64) async_cp16((char*)sY + tid * 16, (const char*)Y + tid * 16);
  wait_async0();
  __syncthreads();

  int nb = lane & 15, mh8 = (lane >> 4) << 3;
  v16bf a1 = load_A_frag(sA[wave], 0, 0, 32);
#pragma unroll
  for (int nt = 0; nt < 3; ++nt) {
    v8f z = {};
    v8f acc = wmma_bf16(a1, load_B_frag_swz(sW1, nt), z);
#pragma unroll
    for (int r = 0; r < 8; ++r) {
      int n = nt * 16 + nb;
      sAct[wave][(mh8 + r) * 64 + n] = (__bf16)ssp(acc[r] + sB1[n]);
    }
  }
  {
    bf16x8 z = {};
    *(bf16x8*)&sAct[wave][(lane >> 1) * 64 + 48 + ((lane & 1) << 3)] = z;
  }

  v8f acc2[4] = {};
#pragma unroll
  for (int ks = 0; ks < 2; ++ks) {
    v16bf a2 = load_A_frag(sAct[wave], 0, ks * 32, 64);
#pragma unroll
    for (int nt = 0; nt < 4; ++nt)
      acc2[nt] = wmma_bf16(a2, load_B_frag_swz(sW2, ks * 4 + nt), acc2[nt]);
  }

  // * Y[a,:] (a = r&3), reduce groups of 4 rows -> (b,i)
  int biBase = (row0 + wave * 16) >> 2, half = lane >> 4;
#pragma unroll
  for (int nt = 0; nt < 4; ++nt) {
    int n = nt * 16 + nb;
    float s0 = 0.f, s1 = 0.f;
#pragma unroll
    for (int r = 0; r < 4; ++r) s0 += acc2[nt][r] * sY[(r & 3) * 64 + n];
#pragma unroll
    for (int r = 4; r < 8; ++r) s1 += acc2[nt][r] * sY[(r & 3) * 64 + n];
    z_nuc[(size_t)(biBase + 2 * half + 0) * KER + n] = s0;
    z_nuc[(size_t)(biBase + 2 * half + 1) * KER + n] = s1;
  }
}

// ---------------------------------------------------------------------------
// Update: z = sum of 3 g-MLPs; x += z. 4 waves x 16 rows / block; branches
// share one LDS weight buffer and accumulate into the same WMMA accumulators.
// ---------------------------------------------------------------------------
__global__ __launch_bounds__(128) void k_g(
    const float* __restrict__ z_same, const float* __restrict__ z_anti,
    const float* __restrict__ z_nuc,
    const __bf16* __restrict__ g1p,   // [3][12 tiles] swizzled: same,anti,nuc
    const float* __restrict__ gb1p,   // [3][96]
    const __bf16* __restrict__ g2p,   // [3][24 tiles] swizzled
    float* __restrict__ x) {
  __shared__ __align__(16) __bf16 sW1[64 * 96];
  __shared__ __align__(16) __bf16 sW2[96 * 128];
  __shared__ __align__(16) float  sB1[96];
  __shared__ __align__(16) __bf16 sZ[4][16 * 64];
  __shared__ __align__(16) __bf16 sAct[4][16 * 96];

  int tid = threadIdx.x, lane = tid & 31, wave = tid >> 5;
  int row0 = blockIdx.x * 64;
  int nb = lane & 15, mh8 = (lane >> 4) << 3;

  v8f accO[8] = {};
  const float* zsrc[3] = {z_same, z_anti, z_nuc};

  for (int br = 0; br < 3; ++br) {
    __syncthreads();  // all waves done with previous branch's weights
    // async weight staging overlapped with z f32->bf16 conversion staging
    for (int t = tid; t < 768; t += 128)
      async_cp16((char*)sW1 + t * 16, (const char*)(g1p + br * 64 * 96) + t * 16);
    for (int t = tid; t < 1536; t += 128)
      async_cp16((char*)sW2 + t * 16, (const char*)(g2p + br * 96 * 128) + t * 16);
    if (tid < 24)
      async_cp16((char*)sB1 + tid * 16, (const char*)(gb1p + br * 96) + tid * 16);
    const float* zp = zsrc[br] + (size_t)row0 * KER;
    for (int c = tid * 8; c < 64 * 64; c += 128 * 8)
      cvt8(&sZ[0][0] + c, zp + c);
    wait_async0();
    __syncthreads();

    // layer 1: K=64 (2 steps), N=96 (6 tiles); padded cols give ssp(0)=0
    v8f acc1[6] = {};
#pragma unroll
    for (int ks = 0; ks < 2; ++ks) {
      v16bf a = load_A_frag(sZ[wave], 0, ks * 32, 64);
#pragma unroll
      for (int nt = 0; nt < 6; ++nt)
        acc1[nt] = wmma_bf16(a, load_B_frag_swz(sW1, ks * 6 + nt), acc1[nt]);
    }
#pragma unroll
    for (int nt = 0; nt < 6; ++nt)
#pragma unroll
      for (int r = 0; r < 8; ++r) {
        int n = nt * 16 + nb;
        sAct[wave][(mh8 + r) * 96 + n] = (__bf16)ssp(acc1[nt][r] + sB1[n]);
      }

    // layer 2: K=96 (3 steps), N=128 (8 tiles); accumulate across branches
#pragma unroll
    for (int ks = 0; ks < 3; ++ks) {
      v16bf a = load_A_frag(sAct[wave], 0, ks * 32, 96);
#pragma unroll
      for (int nt = 0; nt < 8; ++nt)
        accO[nt] = wmma_bf16(a, load_B_frag_swz(sW2, ks * 8 + nt), accO[nt]);
    }
  }

  float* xp = x + (size_t)(row0 + wave * 16) * EMB;
#pragma unroll
  for (int nt = 0; nt < 8; ++nt)
#pragma unroll
    for (int r = 0; r < 8; ++r)
      xp[(mh8 + r) * EMB + nt * 16 + nb] += accO[nt][r];
}

// ---------------------------------------------------------------------------
// Host launcher
// ---------------------------------------------------------------------------
extern "C" void kernel_launch(void* const* d_in, const int* in_sizes, int n_in,
                              void* d_out, int out_size, void* d_ws, size_t ws_size,
                              hipStream_t stream) {
  const float* edges_elec = (const float*)d_in[0];  // [B,32,32,32]
  const float* edges_nuc  = (const float*)d_in[1];  // [B,32,4,32]
  const float* X          = (const float*)d_in[2];  // [32,128]
  const float* Yv         = (const float*)d_in[3];  // [4,64]
  const float* hW         = (const float*)d_in[4];  // [3,128,64]
  float* xout = (float*)d_out;                      // [B,32,128]

  char* ws = (char*)d_ws;
  const size_t N_EDGE = (size_t)NB * NE * NE * BAS;   // 33,554,432
  const size_t N_ENUC = (size_t)NB * NE * NN * BAS;   //  4,194,304
  size_t off = 0;
  __bf16* edges_bf = (__bf16*)(ws + off); off += N_EDGE * 2;
  __bf16* enuc_bf  = (__bf16*)(ws + off); off += N_ENUC * 2;
  float*  hbuf     = (float*)(ws + off);  off += (size_t)NB * NE * KER * 4;
  float*  zs       = (float*)(ws + off);  off += (size_t)NB * NE * KER * 4;
  float*  za       = (float*)(ws + off);  off += (size_t)NB * NE * KER * 4;
  float*  zn       = (float*)(ws + off);  off += (size_t)NB * NE * KER * 4;
  __bf16* hWbf     = (__bf16*)(ws + off); off += 3 * 128 * 64 * 2;
  __bf16* w1p      = (__bf16*)(ws + off); off += 9 * 32 * 48 * 2;   // [n][lab]
  float*  b1p      = (float*)(ws + off);  off += 9 * 48 * 4;
  __bf16* w2p      = (__bf16*)(ws + off); off += 9 * 64 * 64 * 2;
  __bf16* g1p      = (__bf16*)(ws + off); off += 9 * 64 * 96 * 2;
  float*  gb1p     = (float*)(ws + off);  off += 9 * 96 * 4;
  __bf16* g2p      = (__bf16*)(ws + off); off += 9 * 96 * 128 * 2;
  (void)off; (void)ws_size; (void)in_sizes; (void)n_in; (void)out_size;

  // ---- prep: bf16 conversion (x8 vectorized) + weight swizzle/pad ----
  k_cvt8<<<(unsigned)(N_EDGE / 8 / 256), 256, 0, stream>>>(edges_elec, edges_bf, N_EDGE / 8);
  k_cvt8<<<(unsigned)(N_ENUC / 8 / 256), 256, 0, stream>>>(edges_nuc, enuc_bf, N_ENUC / 8);

  for (int n = 0; n < 3; ++n) {
    k_swz_w<<<128, 64, 0, stream>>>(hW + (size_t)n * 128 * 64,
                                    hWbf + (size_t)n * 128 * 64, 128, 64);
    for (int lab = 0; lab < 3; ++lab) {
      const float* w1  = (const float*)d_in[5 + lab * 6 + 0] + (size_t)n * BAS * MW;
      const float* b1  = (const float*)d_in[5 + lab * 6 + 1] + (size_t)n * MW;
      const float* w2  = (const float*)d_in[5 + lab * 6 + 2] + (size_t)n * MW * KER;
      const float* g1  = (const float*)d_in[5 + lab * 6 + 3] + (size_t)n * KER * MG;
      const float* gb1 = (const float*)d_in[5 + lab * 6 + 4] + (size_t)n * MG;
      const float* g2  = (const float*)d_in[5 + lab * 6 + 5] + (size_t)n * MG * EMB;
      int s = n * 3 + lab;
      k_swz_w<<<32, 48, 0, stream>>>(w1, w1p + (size_t)s * 32 * 48, 32, 45);
      k_pad_f32<<<1, 64, 0, stream>>>(b1, b1p + (size_t)s * 48, 45, 48);
      k_swz_w<<<64, 64, 0, stream>>>(w2, w2p + (size_t)s * 64 * 64, 45, 64);
      k_swz_w<<<64, 96, 0, stream>>>(g1, g1p + (size_t)s * 64 * 96, 64, 91);
      k_pad_f32<<<1, 128, 0, stream>>>(gb1, gb1p + (size_t)s * 96, 91, 96);
      k_swz_w<<<96, 128, 0, stream>>>(g2, g2p + (size_t)s * 96 * 128, 91, 128);
    }
  }

  k_xinit<<<(unsigned)((size_t)NB * NE * EMB / 4 / 256), 256, 0, stream>>>(
      (const float4*)X, (float4*)xout);

  // ---- 3 interaction rounds ----
  for (int n = 0; n < 3; ++n) {
    k_h<<<NB * NE / 128, 256, 0, stream>>>(xout, hWbf + (size_t)n * 128 * 64, hbuf);
    k_pair<<<NB * NE / 4, 256, 0, stream>>>(
        edges_bf,
        w1p + (size_t)(n * 3) * 32 * 48,   // same, anti contiguous
        b1p + (size_t)(n * 3) * 48,
        w2p + (size_t)(n * 3) * 64 * 64,
        hbuf, zs, za);
    k_nuc<<<NB * NE * NN / 128, 256, 0, stream>>>(
        enuc_bf,
        w1p + (size_t)(n * 3 + 2) * 32 * 48,
        b1p + (size_t)(n * 3 + 2) * 48,
        w2p + (size_t)(n * 3 + 2) * 64 * 64,
        Yv, zn);
    k_g<<<NB * NE / 64, 128, 0, stream>>>(
        zs, za, zn,
        g1p + (size_t)(n * 3) * 64 * 96,
        gb1p + (size_t)(n * 3) * 96,
        g2p + (size_t)(n * 3) * 96 * 128,
        xout);
  }
}
